// EfficientIGPM_78417512890998
// MI455X (gfx1250) — compile-verified, matrix operand order checked by reading, not run on hardware
//
#include <hip/hip_runtime.h>
#include <hip/hip_bf16.h>
#include <math.h>

// ---------------- problem constants ----------------
#define D_      1536
#define INSTR_  512
#define NSLOT_  8
#define RANK_   64
#define E_      384
#define H_      256
#define CTX_    192
#define COMB_   704
#define SEQ_    8192
#define TOK_    50
#define TOPK_   3
#define FOURH_  1024

// ---------------- workspace layout (float offsets) ----------------
#define WS_POOLED   0        // 1536
#define WS_INSTR    1536     // 512
#define WS_TOPW     2048     // 3
#define WS_TOPI     2052     // 3 (ints)
#define WS_GNORM    2056
#define WS_ALR      2057
#define WS_HU       2768     // 128
#define WS_HV       2896     // 128
#define WS_BUPD     3024     // 1536
#define WS_TRANS    4560     // 1536
#define WS_CORR     6160     // 1536
#define WS_COMBP    7696     // 1536
#define WS_RBIAS    9232     // 1536
#define WS_PLAST    10768    // 3*1536
#define WS_U        15376    // 98304  u[d][r] = u[d*64+r]
#define WS_V        113680   // 98304  v[r][d] = v[r*1536+d]
#define WS_PPART    211984   // 128*1536 pooling partials

__device__ __forceinline__ float sigf(float x) { return 1.0f / (1.0f + __expf(-x)); }

// ================= pooling (deterministic two-pass) =================
__global__ void pool_partial_kernel(const float* __restrict__ A, float* __restrict__ part) {
    int tid = threadIdx.x;                    // 256
    int r0  = blockIdx.x * 64;                // 128 blocks
    for (int c = tid; c < D_; c += 256) {
        float s = 0.f;
        for (int r = 0; r < 64; ++r) s += A[(size_t)(r0 + r) * D_ + c];
        part[(size_t)blockIdx.x * D_ + c] = s;
    }
}

__global__ void pool_reduce_kernel(const float* __restrict__ part, float* __restrict__ pooled) {
    int c = blockIdx.x * 256 + threadIdx.x;
    if (c < D_) {
        float s = 0.f;
        for (int b = 0; b < 128; ++b) s += part[(size_t)b * D_ + c];
        pooled[c] = s * (1.0f / SEQ_);
    }
}

// ================= biLSTM instruction encoder + slot selection =================
__global__ void lstm_select_kernel(const int* __restrict__ tokens, const float* __restrict__ emb,
                                   const float* __restrict__ WiF, const float* __restrict__ WhF,
                                   const float* __restrict__ biF, const float* __restrict__ bhF,
                                   const float* __restrict__ WiB, const float* __restrict__ WhB,
                                   const float* __restrict__ biB, const float* __restrict__ bhB,
                                   const float* __restrict__ selW1, const float* __restrict__ selb1,
                                   const float* __restrict__ selW2, const float* __restrict__ selb2,
                                   float* __restrict__ ws) {
    __shared__ float xcur[E_];
    __shared__ float zsh[FOURH_];
    __shared__ float hsh[H_], csh[H_];
    __shared__ float hsum[INSTR_];
    __shared__ float insh[INSTR_];
    __shared__ float s1[16];
    int tid = threadIdx.x;                    // 1024
    if (tid < INSTR_) hsum[tid] = 0.f;
    if (tid < H_) { hsh[tid] = 0.f; csh[tid] = 0.f; }
    __syncthreads();
    for (int dir = 0; dir < 2; ++dir) {
        const float* Wi = dir ? WiB : WiF;
        const float* Wh = dir ? WhB : WhF;
        const float* bi = dir ? biB : biF;
        const float* bh = dir ? bhB : bhF;
        for (int step = 0; step < TOK_; ++step) {
            int t = dir ? (TOK_ - 1 - step) : step;
            int tok = tokens[t];
            if (tid < E_) xcur[tid] = emb[(size_t)tok * E_ + tid];
            __syncthreads();
            float z = bi[tid] + bh[tid];
            const float* wr = Wi + (size_t)tid * E_;
            for (int j = 0; j < E_; j += 4) {
                float4 w = *(const float4*)(wr + j);
                z += w.x * xcur[j] + w.y * xcur[j + 1] + w.z * xcur[j + 2] + w.w * xcur[j + 3];
            }
            const float* hr = Wh + (size_t)tid * H_;
            for (int j = 0; j < H_; j += 4) {
                float4 w = *(const float4*)(hr + j);
                z += w.x * hsh[j] + w.y * hsh[j + 1] + w.z * hsh[j + 2] + w.w * hsh[j + 3];
            }
            zsh[tid] = z;
            __syncthreads();
            if (tid < H_) {                   // torch gate order i,f,g,o
                float ig = sigf(zsh[tid]);
                float fg = sigf(zsh[tid + H_]);
                float gg = tanhf(zsh[tid + 2 * H_]);
                float og = sigf(zsh[tid + 3 * H_]);
                float c = fg * csh[tid] + ig * gg;
                float h = og * tanhf(c);
                csh[tid] = c; hsh[tid] = h;
                hsum[dir * H_ + tid] += h;
            }
            __syncthreads();
        }
        if (tid < H_) { hsh[tid] = 0.f; csh[tid] = 0.f; }
        __syncthreads();
    }
    if (tid < INSTR_) { insh[tid] = hsum[tid] * (1.0f / TOK_); ws[WS_INSTR + tid] = insh[tid]; }
    __syncthreads();
    if (tid < 16) {
        float s = selb1[tid];
        const float* wr = selW1 + (size_t)tid * INSTR_;
        for (int j = 0; j < INSTR_; ++j) s += wr[j] * insh[j];
        s1[tid] = fmaxf(s, 0.f);
    }
    __syncthreads();
    if (tid == 0) {
        float lg[NSLOT_]; float mx = -1e30f;
        for (int k = 0; k < NSLOT_; ++k) {
            float s = selb2[k];
            for (int j = 0; j < 16; ++j) s += selW2[k * 16 + j] * s1[j];
            lg[k] = s; mx = fmaxf(mx, s);
        }
        float den = 0.f;
        for (int k = 0; k < NSLOT_; ++k) { lg[k] = __expf(lg[k] - mx); den += lg[k]; }
        for (int k = 0; k < NSLOT_; ++k) lg[k] /= den;
        int* topi = (int*)(ws + WS_TOPI);
        for (int k = 0; k < TOPK_; ++k) {
            int bidx = 0; float bv = -1e30f;
            for (int j = 0; j < NSLOT_; ++j) if (lg[j] > bv) { bv = lg[j]; bidx = j; }
            ws[WS_TOPW + k] = bv; topi[k] = bidx; lg[bidx] = -1e30f;
        }
    }
}

// ================= small meta MLPs: ce -> comb -> hu/hv/bh -> b_upd =================
__global__ void meta_small_kernel(const float* __restrict__ ceW1, const float* __restrict__ ceb1,
                                  const float* __restrict__ ceW2, const float* __restrict__ ceb2,
                                  const float* __restrict__ wuW1, const float* __restrict__ wub1,
                                  const float* __restrict__ wvW1, const float* __restrict__ wvb1,
                                  const float* __restrict__ bgW1, const float* __restrict__ bgb1,
                                  const float* __restrict__ bgW2, const float* __restrict__ bgb2,
                                  float* __restrict__ ws) {
    __shared__ float pl[D_];
    __shared__ float ceh[E_];
    __shared__ float comb[COMB_];
    __shared__ float bhid[768];
    int tid = threadIdx.x;                    // 1024
    for (int i = tid; i < D_; i += 1024) pl[i] = ws[WS_POOLED + i];
    __syncthreads();
    if (tid < E_) {
        float s = ceb1[tid];
        const float* wr = ceW1 + (size_t)tid * D_;
        for (int j = 0; j < D_; j += 4) {
            float4 w = *(const float4*)(wr + j);
            s += w.x * pl[j] + w.y * pl[j + 1] + w.z * pl[j + 2] + w.w * pl[j + 3];
        }
        ceh[tid] = fmaxf(s, 0.f);
    }
    __syncthreads();
    if (tid < CTX_) {
        float s = ceb2[tid];
        const float* wr = ceW2 + (size_t)tid * E_;
        for (int j = 0; j < E_; ++j) s += wr[j] * ceh[j];
        comb[tid] = s;
    }
    if (tid >= 256 && tid < 256 + INSTR_) comb[CTX_ + (tid - 256)] = ws[WS_INSTR + (tid - 256)];
    __syncthreads();
    if (tid < 256) {                          // hu / hv
        int i = tid & 127;
        const float* W = (tid < 128) ? wuW1 : wvW1;
        const float* b = (tid < 128) ? wub1 : wvb1;
        float s = b[i];
        const float* wr = W + (size_t)i * COMB_;
        for (int j = 0; j < COMB_; ++j) s += wr[j] * comb[j];
        ws[(tid < 128 ? WS_HU : WS_HV) + i] = fmaxf(s, 0.f);
    }
    if (tid >= 256) {                         // bias-gen hidden (768)
        int i = tid - 256;
        float s = bgb1[i];
        const float* wr = bgW1 + (size_t)i * COMB_;
        for (int j = 0; j < COMB_; ++j) s += wr[j] * comb[j];
        bhid[i] = fmaxf(s, 0.f);
    }
    __syncthreads();
    for (int e = tid; e < D_; e += 1024) {
        float s = bgb2[e];
        const float* wr = bgW2 + (size_t)e * 768;
        for (int j = 0; j < 768; j += 4) {
            float4 w = *(const float4*)(wr + j);
            s += w.x * bhid[j] + w.y * bhid[j + 1] + w.z * bhid[j + 2] + w.w * bhid[j + 3];
        }
        ws[WS_BUPD + e] = s;
    }
}

// ================= u / v generation: wave-per-row 128-dot =================
__global__ void uv_kernel(const float* __restrict__ W2, const float* __restrict__ b2,
                          const float* __restrict__ hws, float* __restrict__ outv) {
    __shared__ float hs[128];
    int tid = threadIdx.x;
    if (tid < 128) hs[tid] = hws[tid];
    __syncthreads();
    int row = blockIdx.x * 8 + (tid >> 5);
    int lane = tid & 31;
    const float* wr = W2 + (size_t)row * 128;
    float s = 0.f;
#pragma unroll
    for (int j = 0; j < 128; j += 32) s += wr[j + lane] * hs[j + lane];
#pragma unroll
    for (int off = 16; off > 0; off >>= 1) s += __shfl_xor(s, off, 32);
    if (lane == 0) outv[row] = s + b2[row];
}

// ================= ||u@v||_F via Gram matrices (no D*D materialization) =================
__global__ void gram_kernel(const float* __restrict__ scale, float* __restrict__ ws) {
    __shared__ float red[1024];
    int tid = threadIdx.x;
    const float* u = ws + WS_U;
    const float* v = ws + WS_V;
    float tot = 0.f;
    for (int p = tid; p < RANK_ * RANK_; p += 1024) {
        int r = p >> 6, c = p & 63;
        float gu = 0.f, gv = 0.f;
        for (int d = 0; d < D_; ++d) gu += u[d * RANK_ + r] * u[d * RANK_ + c];
        const float* vr = v + (size_t)r * D_;
        const float* vc = v + (size_t)c * D_;
        for (int e = 0; e < D_; ++e) gv += vr[e] * vc[e];
        tot += gu * gv;
    }
    red[tid] = tot; __syncthreads();
    for (int st = 512; st > 0; st >>= 1) { if (tid < st) red[tid] += red[tid + st]; __syncthreads(); }
    if (tid == 0) {
        float sc = scale[0];
        float fro2 = 0.01f * sc * sc * red[0];    // wg = 0.1*scale*(u@v)
        float gn = sqrtf(fmaxf(fro2, 0.f));
        ws[WS_GNORM] = gn;
        ws[WS_ALR] = 0.01f * fminf(fmaxf(gn, 0.1f), 2.0f);
    }
}

// ================= generic wave-per-row matvec (1536x1536) =================
__global__ void matvec_rows_kernel(const float* __restrict__ W, const float* __restrict__ x,
                                   const float* __restrict__ b, float* __restrict__ y, int do_relu) {
    __shared__ float xs[D_];
    int tid = threadIdx.x;
    for (int i = tid; i < D_; i += 256) xs[i] = x[i];
    __syncthreads();
    int row = blockIdx.x * 8 + (tid >> 5);
    int lane = tid & 31;
    const float* wr = W + (size_t)row * D_;
    float s = 0.f;
    for (int c = lane * 4; c < D_; c += 128) {
        float4 w = *(const float4*)(wr + c);
        s += w.x * xs[c] + w.y * xs[c + 1] + w.z * xs[c + 2] + w.w * xs[c + 3];
    }
#pragma unroll
    for (int off = 16; off > 0; off >>= 1) s += __shfl_xor(s, off, 32);
    if (lane == 0) {
        float r = s + b[row];
        y[row] = do_relu ? fmaxf(r, 0.f) : r;
    }
}

// ================= low-rank correction + effect scalar =================
__global__ void finalize_small_kernel(const float* __restrict__ scale, const float* __restrict__ ps_all,
                                      float* __restrict__ ws, float* __restrict__ d_out) {
    __shared__ float ts[D_];
    __shared__ float vt[RANK_];
    int tid = threadIdx.x;                    // 256
    for (int i = tid; i < D_; i += 256) ts[i] = ws[WS_TRANS + i];
    __syncthreads();
    if (tid < RANK_) {
        const float* vr = ws + WS_V + (size_t)tid * D_;
        float s = 0.f;
        for (int e = 0; e < D_; ++e) s += vr[e] * ts[e];
        vt[tid] = s;
    }
    __syncthreads();
    float alr = ws[WS_ALR];
    float sc = scale[0];
    for (int e = tid; e < D_; e += 256) {
        const float* ur = ws + WS_U + (size_t)e * RANK_;
        float s = 0.f;
#pragma unroll
        for (int r = 0; r < RANK_; ++r) s += ur[r] * vt[r];
        // alr * (0.1*scale*(u@v)@t + 0.1*b_upd)
        ws[WS_CORR + e] = alr * 0.1f * (sc * s + ws[WS_BUPD + e]);
    }
    if (tid == 0) {
        const int* topi = (const int*)(ws + WS_TOPI);
        float gn = ws[WS_GNORM];
        float eff = 0.f;
        for (int k = 0; k < TOPK_; ++k) {
            float w = ws[WS_TOPW + k];
            float we = (w > 0.1f) ? w : 0.f;
            float ps = 0.99f * ps_all[topi[k]] + 0.01f * gn;
            eff += we * fminf(fmaxf(ps, 0.01f), 1.0f);
        }
        d_out[(size_t)SEQ_ * D_] = eff;       // second tuple output
    }
}

// ================= per-slot fast-weight matvec (28 MB stream) =================
__global__ void plastic_kernel(const float* __restrict__ fw_all, const float* __restrict__ fb_all,
                               float* __restrict__ ws) {
    __shared__ float ts[D_];
    int tid = threadIdx.x;
    for (int i = tid; i < D_; i += 256) ts[i] = ws[WS_TRANS + i];
    __syncthreads();
    int row = blockIdx.x * 8 + (tid >> 5);    // [0, 3*1536)
    int lane = tid & 31;
    int k = row / D_;
    int e = row - k * D_;
    const int* topi = (const int*)(ws + WS_TOPI);
    int ti = topi[k];
    const float* wr = fw_all + ((size_t)ti * D_ + e) * D_;
    float s = 0.f;
    for (int c = lane * 4; c < D_; c += 128) {
        float4 w = *(const float4*)(wr + c);
        s += w.x * ts[c] + w.y * ts[c + 1] + w.z * ts[c + 2] + w.w * ts[c + 3];
    }
#pragma unroll
    for (int off = 16; off > 0; off >>= 1) s += __shfl_xor(s, off, 32);
    if (lane == 0)
        ws[WS_PLAST + row] = s + fb_all[(size_t)ti * D_ + e] + ws[WS_CORR + e];
}

// ================= layernorm + weighted combine =================
__global__ void combine_kernel(const float* __restrict__ lng, const float* __restrict__ lnb,
                               float* __restrict__ ws) {
    __shared__ float red[1024];
    __shared__ float stats[2];
    int tid = threadIdx.x;                    // 1024
    float acc0 = 0.f, acc1 = 0.f;
    for (int k = 0; k < TOPK_; ++k) {
        float w = ws[WS_TOPW + k];
        float we = (w > 0.1f) ? w : 0.f;
        const float* pk = ws + WS_PLAST + k * D_;
        float a = 0.f, b = 0.f;
        for (int e = tid; e < D_; e += 1024) { float p = pk[e]; a += p; b += p * p; }
        red[tid] = a; __syncthreads();
        for (int st = 512; st > 0; st >>= 1) { if (tid < st) red[tid] += red[tid + st]; __syncthreads(); }
        if (tid == 0) stats[0] = red[0] * (1.0f / D_);
        __syncthreads();
        red[tid] = b; __syncthreads();
        for (int st = 512; st > 0; st >>= 1) { if (tid < st) red[tid] += red[tid + st]; __syncthreads(); }
        if (tid == 0) {
            float m = stats[0];
            float var = red[0] * (1.0f / D_) - m * m;
            stats[1] = rsqrtf(var + 1e-5f);
        }
        __syncthreads();
        float m = stats[0], rs = stats[1];
        if (tid < D_)        acc0 += we * ((pk[tid] - m) * rs * lng[tid] + lnb[tid]);
        if (tid + 1024 < D_) acc1 += we * ((pk[tid + 1024] - m) * rs * lng[tid + 1024] + lnb[tid + 1024]);
        __syncthreads();
    }
    if (tid < D_)        ws[WS_COMBP + tid] = acc0;
    if (tid + 1024 < D_) ws[WS_COMBP + tid + 1024] = acc1;
}

// ================= the big GEMM: out = hidden @ out_W^T + row_bias =================
// bf16 WMMA (f32 accumulate), double-buffered LDS, static per-thread staging roles.
typedef __attribute__((ext_vector_type(16))) __bf16 v16bf;
typedef __attribute__((ext_vector_type(8)))  float  v8f;
union FragU { v16bf v; unsigned long long q[4]; };

// round-half-up f32->bf16 pair packed via one v_perm_b32 (bytes {2,3} of lo, {2,3} of hi)
__device__ __forceinline__ unsigned int pack_bf2(float a, float b) {
    unsigned int ua = __float_as_uint(a) + 0x8000u;
    unsigned int ub = __float_as_uint(b) + 0x8000u;
    return __builtin_amdgcn_perm(ub, ua, 0x07060302u);
}

#define BM 128
#define BN 64
#define BK 32
#define LDA 36   // padded bf16 row stride (72B) -> conflict-free ds_load_b64 gathers

__global__ __launch_bounds__(256) void gemm_out_kernel(const float* __restrict__ A,
                                                       const float* __restrict__ W,
                                                       const float* __restrict__ rbias,
                                                       float* __restrict__ out) {
    __shared__ __align__(16) unsigned short As[2][BM * LDA];
    __shared__ __align__(16) unsigned short Bs[2][BN * LDA];
    __shared__ float BiasS[BN];
    int tid = threadIdx.x;
    int bm = blockIdx.y * BM;
    int bn = blockIdx.x * BN;
    if (tid < BN) BiasS[tid] = rbias[bn + tid];
    int lane = tid & 31;
    int wave = tid >> 5;
    int wmg = wave & 3;                       // 32-row macro tile
    int wng = wave >> 2;                      // 32-col macro tile
    int l  = lane & 15;
    int hg = lane >> 4;
    // 16-bit A-matrix K-chunk swizzle (lanes 0-15: K {0-7,16-23}; 16-31: {8-15,24-31})
    int kc0 = 8 * hg, kc1 = 4 + 8 * hg, kc2 = 16 + 8 * hg, kc3 = 20 + 8 * hg;
    // static staging roles: 4 A float4-chunks + 2 B float4-chunks per thread
    const float* aglb[4]; int aoff[4];
    const float* bglb[2]; int boff[2];
#pragma unroll
    for (int j = 0; j < 4; ++j) {
        int q = tid + j * 256, row = q >> 3, c4 = q & 7;
        aglb[j] = A + (size_t)(bm + row) * D_ + c4 * 4;
        aoff[j] = row * LDA + c4 * 4;
    }
#pragma unroll
    for (int j = 0; j < 2; ++j) {
        int q = tid + j * 256, row = q >> 3, c4 = q & 7;
        bglb[j] = W + (size_t)(bn + row) * D_ + c4 * 4;
        boff[j] = row * LDA + c4 * 4;
    }
    float4 areg[4], breg[2];
    // prologue: stage tile 0
#pragma unroll
    for (int j = 0; j < 4; ++j) areg[j] = *(const float4*)(aglb[j]);
#pragma unroll
    for (int j = 0; j < 2; ++j) breg[j] = *(const float4*)(bglb[j]);
#pragma unroll
    for (int j = 0; j < 4; ++j) {
        unsigned int* dp = (unsigned int*)&As[0][aoff[j]];
        dp[0] = pack_bf2(areg[j].x, areg[j].y);
        dp[1] = pack_bf2(areg[j].z, areg[j].w);
    }
#pragma unroll
    for (int j = 0; j < 2; ++j) {
        unsigned int* dp = (unsigned int*)&Bs[0][boff[j]];
        dp[0] = pack_bf2(breg[j].x, breg[j].y);
        dp[1] = pack_bf2(breg[j].z, breg[j].w);
    }
    __syncthreads();
    v8f acc00 = {}, acc01 = {}, acc10 = {}, acc11 = {};
    int par = 0;
    for (int k0 = 0; k0 < D_; k0 += BK) {
        int nk = k0 + BK;
        bool more = nk < D_;
        if (more) {                            // phase 1: issue next-tile global loads
#pragma unroll
            for (int j = 0; j < 4; ++j) {
                areg[j] = *(const float4*)(aglb[j] + nk);
                if (nk + BK < D_) __builtin_prefetch(aglb[j] + nk + BK, 0, 1); // global_prefetch_b8
            }
#pragma unroll
            for (int j = 0; j < 2; ++j) {
                breg[j] = *(const float4*)(bglb[j] + nk);
                if (nk + BK < D_) __builtin_prefetch(bglb[j] + nk + BK, 0, 1);
            }
        }
        // phase 2: compute on current LDS buffer (overlaps next-tile global latency)
        {
            FragU a0, a1, b0, b1;
            const unsigned short* ar0 = &As[par][(wmg * 32 + l) * LDA];
            const unsigned short* ar1 = &As[par][(wmg * 32 + 16 + l) * LDA];
            const unsigned short* br0 = &Bs[par][(wng * 32 + l) * LDA];
            const unsigned short* br1 = &Bs[par][(wng * 32 + 16 + l) * LDA];
            a0.q[0] = *(const unsigned long long*)(ar0 + kc0);
            a0.q[1] = *(const unsigned long long*)(ar0 + kc1);
            a0.q[2] = *(const unsigned long long*)(ar0 + kc2);
            a0.q[3] = *(const unsigned long long*)(ar0 + kc3);
            a1.q[0] = *(const unsigned long long*)(ar1 + kc0);
            a1.q[1] = *(const unsigned long long*)(ar1 + kc1);
            a1.q[2] = *(const unsigned long long*)(ar1 + kc2);
            a1.q[3] = *(const unsigned long long*)(ar1 + kc3);
            b0.q[0] = *(const unsigned long long*)(br0 + kc0);
            b0.q[1] = *(const unsigned long long*)(br0 + kc1);
            b0.q[2] = *(const unsigned long long*)(br0 + kc2);
            b0.q[3] = *(const unsigned long long*)(br0 + kc3);
            b1.q[0] = *(const unsigned long long*)(br1 + kc0);
            b1.q[1] = *(const unsigned long long*)(br1 + kc1);
            b1.q[2] = *(const unsigned long long*)(br1 + kc2);
            b1.q[3] = *(const unsigned long long*)(br1 + kc3);
            acc00 = __builtin_amdgcn_wmma_f32_16x16x32_bf16(false, a0.v, false, b0.v, (short)0, acc00, false, false);
            acc01 = __builtin_amdgcn_wmma_f32_16x16x32_bf16(false, a0.v, false, b1.v, (short)0, acc01, false, false);
            acc10 = __builtin_amdgcn_wmma_f32_16x16x32_bf16(false, a1.v, false, b0.v, (short)0, acc10, false, false);
            acc11 = __builtin_amdgcn_wmma_f32_16x16x32_bf16(false, a1.v, false, b1.v, (short)0, acc11, false, false);
        }
        if (more) {                            // phase 3: convert + store into other buffer
            int np = par ^ 1;
#pragma unroll
            for (int j = 0; j < 4; ++j) {
                unsigned int* dp = (unsigned int*)&As[np][aoff[j]];
                dp[0] = pack_bf2(areg[j].x, areg[j].y);
                dp[1] = pack_bf2(areg[j].z, areg[j].w);
            }
#pragma unroll
            for (int j = 0; j < 2; ++j) {
                unsigned int* dp = (unsigned int*)&Bs[np][boff[j]];
                dp[0] = pack_bf2(breg[j].x, breg[j].y);
                dp[1] = pack_bf2(breg[j].z, breg[j].w);
            }
        }
        __syncthreads();                       // one barrier per K-step
        par ^= 1;
    }
    // C layout: VGPR r -> (lanes 0-15: M=r, lanes 16-31: M=8+r), N = lane&15
    int gm0 = bm + wmg * 32 + hg * 8;
    int gc0 = bn + wng * 32 + l;
    float bias0 = BiasS[wng * 32 + l];
    float bias1 = BiasS[wng * 32 + 16 + l];
#pragma unroll
    for (int r = 0; r < 8; ++r) {
        out[(size_t)(gm0 + r) * D_ + gc0]           = acc00[r] + bias0;
        out[(size_t)(gm0 + r) * D_ + gc0 + 16]      = acc01[r] + bias1;
        out[(size_t)(gm0 + 16 + r) * D_ + gc0]      = acc10[r] + bias0;
        out[(size_t)(gm0 + 16 + r) * D_ + gc0 + 16] = acc11[r] + bias1;
    }
}

// ================= host launcher =================
extern "C" void kernel_launch(void* const* d_in, const int* in_sizes, int n_in,
                              void* d_out, int out_size, void* d_ws, size_t ws_size,
                              hipStream_t stream) {
    const float* hidden = (const float*)d_in[0];
    const int*   tokens = (const int*)d_in[1];
    const float* emb    = (const float*)d_in[2];
    const float* WiF = (const float*)d_in[3];
    const float* WhF = (const float*)d_in[4];
    const float* biF = (const float*)d_in[5];
    const float* bhF = (const float*)d_in[6];
    const float* WiB = (const float*)d_in[7];
    const float* WhB = (const float*)d_in[8];
    const float* biB = (const float*)d_in[9];
    const float* bhB = (const float*)d_in[10];
    const float* ceW1 = (const float*)d_in[11];
    const float* ceb1 = (const float*)d_in[12];
    const float* ceW2 = (const float*)d_in[13];
    const float* ceb2 = (const float*)d_in[14];
    const float* wuW1 = (const float*)d_in[15];
    const float* wub1 = (const float*)d_in[16];
    const float* wuW2 = (const float*)d_in[17];
    const float* wub2 = (const float*)d_in[18];
    const float* wvW1 = (const float*)d_in[19];
    const float* wvb1 = (const float*)d_in[20];
    const float* wvW2 = (const float*)d_in[21];
    const float* wvb2 = (const float*)d_in[22];
    const float* bgW1 = (const float*)d_in[23];
    const float* bgb1 = (const float*)d_in[24];
    const float* bgW2 = (const float*)d_in[25];
    const float* bgb2 = (const float*)d_in[26];
    const float* scale = (const float*)d_in[27];
    const float* trW = (const float*)d_in[28];
    const float* trb = (const float*)d_in[29];
    const float* lng = (const float*)d_in[30];
    const float* lnb = (const float*)d_in[31];
    const float* fw_all = (const float*)d_in[32];
    const float* fb_all = (const float*)d_in[33];
    const float* ps_all = (const float*)d_in[34];
    const float* selW1 = (const float*)d_in[35];
    const float* selb1 = (const float*)d_in[36];
    const float* selW2 = (const float*)d_in[37];
    const float* selb2 = (const float*)d_in[38];
    const float* outW = (const float*)d_in[39];
    const float* outb = (const float*)d_in[40];
    float* ws  = (float*)d_ws;
    float* out = (float*)d_out;

    pool_partial_kernel<<<SEQ_ / 64, 256, 0, stream>>>(hidden, ws + WS_PPART);
    pool_reduce_kernel<<<(D_ + 255) / 256, 256, 0, stream>>>(ws + WS_PPART, ws + WS_POOLED);
    lstm_select_kernel<<<1, 1024, 0, stream>>>(tokens, emb, WiF, WhF, biF, bhF,
                                               WiB, WhB, biB, bhB, selW1, selb1, selW2, selb2, ws);
    meta_small_kernel<<<1, 1024, 0, stream>>>(ceW1, ceb1, ceW2, ceb2, wuW1, wub1,
                                              wvW1, wvb1, bgW1, bgb1, bgW2, bgb2, ws);
    uv_kernel<<<(D_ * RANK_) / 8, 256, 0, stream>>>(wuW2, wub2, ws + WS_HU, ws + WS_U);
    uv_kernel<<<(D_ * RANK_) / 8, 256, 0, stream>>>(wvW2, wvb2, ws + WS_HV, ws + WS_V);
    gram_kernel<<<1, 1024, 0, stream>>>(scale, ws);
    matvec_rows_kernel<<<D_ / 8, 256, 0, stream>>>(trW, ws + WS_POOLED, trb, ws + WS_TRANS, 1);
    finalize_small_kernel<<<1, 256, 0, stream>>>(scale, ps_all, ws, out);
    plastic_kernel<<<(TOPK_ * D_) / 8, 256, 0, stream>>>(fw_all, fb_all, ws);
    combine_kernel<<<1, 1024, 0, stream>>>(lng, lnb, ws);
    matvec_rows_kernel<<<D_ / 8, 256, 0, stream>>>(outW, ws + WS_COMBP, outb, ws + WS_RBIAS, 0);
    gemm_out_kernel<<<dim3(D_ / BN, SEQ_ / BM), 256, 0, stream>>>(hidden, outW, ws + WS_RBIAS, out);
}
